// SeqAttention_4733053960342
// MI455X (gfx1250) — matrix operand order repaired and checked by reading, not compiled, not run on hardware
//
#include <hip/hip_runtime.h>
#include <hip/hip_bf16.h>

// ---------------------------------------------------------------------------
// Problem constants (from reference): B=16, S=2048, D=1024, E2=4096
// ---------------------------------------------------------------------------
constexpr int B_  = 16;
constexpr int S_  = 2048;
constexpr int D_  = 1024;
constexpr int E2_ = 4096;
constexpr int KT_ = 128;                 // key tile for attention
constexpr float SCALE_ = 0.03125f;       // 1/sqrt(1024)
constexpr float EPS_ = 1e-5f;

typedef __attribute__((ext_vector_type(16))) __bf16 v16bf;
typedef __attribute__((ext_vector_type(8)))  float  v8f;
typedef __attribute__((ext_vector_type(4)))  unsigned int v4u;

union frag_u { v4u u[2]; v16bf b; };

// ---------------------------------------------------------------------------
// WMMA helpers (CDNA5 wave32 layouts, ISA 7.12.2)
// ---------------------------------------------------------------------------
static __device__ __forceinline__ v8f wmma_bf16(v16bf a, v16bf b, v8f c) {
  // D = A(16x32 bf16) * B(32x16 bf16) + C(16x16 f32)
  return __builtin_amdgcn_wmma_f32_16x16x32_bf16(
      /*neg_a=*/false, a, /*neg_b=*/false, b,
      /*c_mod=*/(short)0, c, /*reuse_a=*/false, /*reuse_b=*/false);
}

// A fragment: 16x32 (MxK), row-major source, leading dim ld (elements).
// Per-lane data = two contiguous 8-element runs -> two b128 loads.
static __device__ __forceinline__ v16bf load_a_frag(const __bf16* src, int ld) {
  const int lane = threadIdx.x & 31;
  const int m = lane & 15;
  const int h = lane >> 4;
  const __bf16* p = src + (size_t)m * ld + 8 * h;
  frag_u f;
  f.u[0] = *(const v4u*)(p);        // K = 8h .. 8h+7
  f.u[1] = *(const v4u*)(p + 16);   // K = 16+8h .. 16+8h+7
  return f.b;
}

// B fragment: 32x16 (KxN), element (k,n) = srcT[n*ld + k] (N-major source).
// Per-lane data = two contiguous 8-element runs -> two b128 loads.
static __device__ __forceinline__ v16bf load_b_frag_t(const __bf16* srcT, int ld) {
  const int lane = threadIdx.x & 31;
  const int n = lane & 15;
  const int h = lane >> 4;
  const __bf16* p = srcT + (size_t)n * ld + 8 * h;
  frag_u f;
  f.u[0] = *(const v4u*)(p);        // K = 8h .. 8h+7
  f.u[1] = *(const v4u*)(p + 16);   // K = 16+8h .. 16+8h+7
  return f.b;
}

// ---------------------------------------------------------------------------
// Kernel 1: f32 -> bf16 cast (grid-stride)
// ---------------------------------------------------------------------------
__global__ void cast_bf16_kernel(const float* __restrict__ src,
                                 __bf16* __restrict__ dst, size_t n) {
  size_t i = (size_t)blockIdx.x * blockDim.x + threadIdx.x;
  size_t stride = (size_t)gridDim.x * blockDim.x;
  for (; i < n; i += stride) dst[i] = (__bf16)src[i];
}

// ---------------------------------------------------------------------------
// Kernel 2: transpose + cast:  src[R][C] f32 -> dst[C][R] bf16
// ---------------------------------------------------------------------------
__global__ void transpose_cast_kernel(const float* __restrict__ src,
                                      __bf16* __restrict__ dst, int R, int C) {
  size_t i = (size_t)blockIdx.x * blockDim.x + threadIdx.x;
  size_t n = (size_t)R * C;
  if (i < n) {
    int r = (int)(i / C);
    int c = (int)(i % C);
    dst[(size_t)c * R + r] = (__bf16)src[i];
  }
}

// Batched version: src[b][R][C] f32 -> dst[b][C][R] bf16 (blockIdx.y = batch)
__global__ void transpose_cast_batched_kernel(const float* __restrict__ src,
                                              __bf16* __restrict__ dst,
                                              int R, int C) {
  const size_t boff = (size_t)blockIdx.y * R * C;
  size_t i = (size_t)blockIdx.x * blockDim.x + threadIdx.x;
  size_t n = (size_t)R * C;
  if (i < n) {
    int r = (int)(i / C);
    int c = (int)(i % C);
    dst[boff + (size_t)c * R + r] = (__bf16)src[boff + i];
  }
}

// ---------------------------------------------------------------------------
// Kernel 3: flash attention (online softmax) + residual + LayerNorm
// 1 workgroup (256 thr, 8 waves) handles 16 query rows of one batch.
// Wave w owns output dim slice [128w, 128w+128) -> 8 f32 C-fragments.
// ---------------------------------------------------------------------------
__global__ __launch_bounds__(256) void attn_ln_kernel(
    const float*  __restrict__ lags,   // [B,S,D] f32 (residual)
    const __bf16* __restrict__ lagsb,  // [B,S,D] bf16 (Q,K)
    const __bf16* __restrict__ lagsT,  // [B,D,S] bf16 (V transposed)
    const float*  __restrict__ g1, const float* __restrict__ be1,
    float*  __restrict__ xf,           // [B,S,D] f32 out
    __bf16* __restrict__ xb)           // [B,S,D] bf16 out
{
  extern __shared__ char smem_raw[];
  float*  sc   = (float*)smem_raw;        // [16][KT_]   score tile (8 KB)
  float*  xbuf = sc + 16 * KT_;           // [16][D_]    x = O + residual (64 KB)
  float*  red  = xbuf + 16 * D_;          // [16][16]    reduction scratch (1 KB)
  float*  mrow = red + 256;               // [16] running max
  float*  lrow = mrow + 16;               // [16] running denom
  float*  crow = lrow + 16;               // [16] correction factor
  __bf16* qs   = (__bf16*)(crow + 16);    // [16][D_]    Q tile bf16 (32 KB)
  __bf16* pb   = qs + 16 * D_;            // [16][KT_]   exp(scores) bf16 (4 KB)

  const int b    = blockIdx.y;
  const int q0   = blockIdx.x * 16;
  const int tid  = threadIdx.x;
  const int w    = tid >> 5;
  const int lane = tid & 31;
  const int nn   = lane & 15;
  const int hh   = lane >> 4;
  const int rrow = tid >> 4;   // 0..15 (reduction row)
  const int rsub = tid & 15;   // 0..15 (reduction slice)

  const __bf16* KB = lagsb + (size_t)b * S_ * D_;
  const __bf16* VT = lagsT + (size_t)b * D_ * S_;

  // stage Q tile (16 rows, contiguous) into LDS with b128 copies
  {
    const v4u* s = (const v4u*)(lagsb + ((size_t)b * S_ + q0) * D_);
    v4u* d = (v4u*)qs;
    for (int i = tid; i < 16 * D_ / 8; i += 256) d[i] = s[i];
  }
  if (tid < 16) { mrow[tid] = -1e30f; lrow[tid] = 0.0f; }
  __syncthreads();

  v8f acc[8];
#pragma unroll
  for (int j = 0; j < 8; ++j) acc[j] = (v8f){};

  for (int kt = 0; kt < S_; kt += KT_) {
    // ---- Phase A: score tile 16 x KT_; wave w computes keys [16w,16w+16) ----
    const __bf16* Kt = KB + (size_t)(kt + 16 * w) * D_;
    v8f s = {};
    for (int kd = 0; kd < D_; kd += 32) {
      __builtin_prefetch(Kt + kd + 256, 0, 1);
      v16bf a  = load_a_frag(qs + kd, D_);    // ds_load_b128 x2
      v16bf bb = load_b_frag_t(Kt + kd, D_);  // K rows row-major == K^T N-major
      s = wmma_bf16(a, bb, s);
    }
#pragma unroll
    for (int r = 0; r < 8; ++r)
      sc[(r + 8 * hh) * KT_ + 16 * w + nn] = s[r] * SCALE_;
    __syncthreads();

    // ---- parallel online-softmax stats: 16 threads/row x 8 elems ----
    {
      const float* p = sc + rrow * KT_ + rsub * 8;
      float m = p[0];
#pragma unroll
      for (int k = 1; k < 8; ++k) m = fmaxf(m, p[k]);
      red[rrow * 16 + rsub] = m;
    }
    __syncthreads();
    if (tid < 16) {
      float m_old = mrow[tid];
      float tmax  = m_old;
      for (int k = 0; k < 16; ++k) tmax = fmaxf(tmax, red[tid * 16 + k]);
      crow[tid] = __expf(m_old - tmax);
      mrow[tid] = tmax;
    }
    __syncthreads();
    {
      const float tmax = mrow[rrow];
      const float* p = sc + rrow * KT_ + rsub * 8;
      __bf16* q = pb + rrow * KT_ + rsub * 8;
      float ssum = 0.0f;
#pragma unroll
      for (int k = 0; k < 8; ++k) {
        float e = __expf(p[k] - tmax);
        ssum += e;
        q[k] = (__bf16)e;
      }
      red[rrow * 16 + rsub] = ssum;
    }
    __syncthreads();
    if (tid < 16) {
      float ssum = 0.0f;
      for (int k = 0; k < 16; ++k) ssum += red[tid * 16 + k];
      lrow[tid] = lrow[tid] * crow[tid] + ssum;
    }

    // ---- rescale accumulators by correction factor ----
#pragma unroll
    for (int r = 0; r < 8; ++r) {
      float f = crow[r + 8 * hh];
#pragma unroll
      for (int j = 0; j < 8; ++j) acc[j][r] *= f;
    }

    // ---- Phase B: O_w += P(16 x KT_) * V(KT_ x 128-slice) ----
    for (int j = 0; j < 8; ++j) {
      const int n0 = 128 * w + 16 * j;
      const __bf16* Vt = VT + (size_t)n0 * S_ + kt;
      for (int kk = 0; kk < KT_; kk += 32) {
        v16bf a  = load_a_frag(pb + kk, KT_);   // ds_load_b128 x2
        v16bf bb = load_b_frag_t(Vt + kk, S_);
        acc[j] = wmma_bf16(a, bb, acc[j]);
      }
    }
    __syncthreads();   // protect sc/pb/red before next tile
  }

  // ---- finalize: x = O/l + residual ----
#pragma unroll
  for (int r = 0; r < 8; ++r) {
    const int row = r + 8 * hh;
    const float inv = 1.0f / lrow[row];
    const float* lrg = lags + ((size_t)b * S_ + q0 + row) * D_;
#pragma unroll
    for (int j = 0; j < 8; ++j) {
      const int col = 128 * w + 16 * j + nn;
      xbuf[row * D_ + col] = acc[j][r] * inv + lrg[col];
    }
  }
  __syncthreads();

  // ---- parallel LayerNorm: 16 threads/row x 64 elems ----
  {
    const float* p = xbuf + rrow * D_ + rsub * 64;
    float ssum = 0.0f;
    for (int k = 0; k < 64; ++k) ssum += p[k];
    red[rrow * 16 + rsub] = ssum;
  }
  __syncthreads();
  if (tid < 16) {
    float mu = 0.0f;
    for (int k = 0; k < 16; ++k) mu += red[tid * 16 + k];
    mrow[tid] = mu * (1.0f / D_);
  }
  __syncthreads();
  {
    const float mu = mrow[rrow];
    const float* p = xbuf + rrow * D_ + rsub * 64;
    float ssum = 0.0f;
    for (int k = 0; k < 64; ++k) { float d = p[k] - mu; ssum += d * d; }
    red[rrow * 16 + rsub] = ssum;
  }
  __syncthreads();
  if (tid < 16) {
    float var = 0.0f;
    for (int k = 0; k < 16; ++k) var += red[tid * 16 + k];
    lrow[tid] = rsqrtf(var * (1.0f / D_) + EPS_);
  }
  __syncthreads();

  for (int i = tid; i < 16 * D_; i += 256) {
    const int row = i >> 10;        // / D_
    const int col = i & (D_ - 1);
    float v = (xbuf[i] - mrow[row]) * lrow[row] * g1[col] + be1[col];
    size_t gi = ((size_t)b * S_ + q0 + row) * D_ + col;
    xf[gi] = v;
    xb[gi] = (__bf16)v;
  }
}

// ---------------------------------------------------------------------------
// Kernel 4: fused FFN (GEMM1 + GELU -> LDS, GEMM2 + residual + LN -> out)
// 1 workgroup (256 thr, 8 waves) handles 16 rows of [B*S, D].
// ---------------------------------------------------------------------------
__global__ __launch_bounds__(256) void ffn_ln_kernel(
    const float*  __restrict__ xf,   // [BS,D] f32 (residual)
    const __bf16* __restrict__ xb,   // [BS,D] bf16
    const __bf16* __restrict__ W1T,  // [E2][D] bf16 (transposed)
    const float*  __restrict__ b1,   // [E2]
    const __bf16* __restrict__ W2T,  // [D][E2] bf16 (transposed)
    const float*  __restrict__ b2,   // [D]
    const float*  __restrict__ g3, const float* __restrict__ be3,
    float* __restrict__ out)         // [BS,D] f32
{
  extern __shared__ char smem_raw[];
  __bf16* hs   = (__bf16*)smem_raw;          // [16][E2_] gelu(x@W1+b1), 128 KB
  float*  fbuf = (float*)(hs + 16 * E2_);    // [16][D_]  ff + residual,  64 KB
  float*  red  = fbuf + 16 * D_;             // [16][16]  reduction scratch
  float*  mu_s = red + 256;                  // [16]
  float*  rs_s = mu_s + 16;                  // [16]
  __bf16* xs   = (__bf16*)(rs_s + 16);       // [16][D_]  x tile bf16, 32 KB

  const int row0 = blockIdx.x * 16;
  const int tid  = threadIdx.x;
  const int w    = tid >> 5;
  const int lane = tid & 31;
  const int nn   = lane & 15;
  const int hh   = lane >> 4;
  const int rrow = tid >> 4;
  const int rsub = tid & 15;

  // stage x tile (16 contiguous rows) into LDS with b128 copies
  {
    const v4u* s = (const v4u*)(xb + (size_t)row0 * D_);
    v4u* d = (v4u*)xs;
    for (int i = tid; i < 16 * D_ / 8; i += 256) d[i] = s[i];
  }
  __syncthreads();

  // ---- Phase 1: wave w computes h cols [512w, 512w+512) ----
  for (int j = 0; j < 32; ++j) {
    const int n0 = 512 * w + 16 * j;
    const __bf16* Wt = W1T + (size_t)n0 * D_;
    v8f accp = {};
    for (int kd = 0; kd < D_; kd += 32) {
      __builtin_prefetch(Wt + kd + 256, 0, 1);
      v16bf a  = load_a_frag(xs + kd, D_);    // ds_load_b128 x2
      v16bf bb = load_b_frag_t(Wt + kd, D_);
      accp = wmma_bf16(a, bb, accp);
    }
    const float bias = b1[n0 + nn];
#pragma unroll
    for (int r = 0; r < 8; ++r) {
      float v = accp[r] + bias;
      float g = 0.5f * v * (1.0f + erff(v * 0.70710678118f));  // exact GELU
      hs[(r + 8 * hh) * E2_ + n0 + nn] = (__bf16)g;
    }
  }
  __syncthreads();

  // ---- Phase 2: wave w computes ff cols [128w, 128w+128) ----
  for (int j = 0; j < 8; ++j) {
    const int n0 = 128 * w + 16 * j;
    const __bf16* Wt = W2T + (size_t)n0 * E2_;
    v8f accp = {};
    for (int kd = 0; kd < E2_; kd += 32) {
      __builtin_prefetch(Wt + kd + 256, 0, 1);
      v16bf a  = load_a_frag(hs + kd, E2_);   // ds_load_b128 x2
      v16bf bb = load_b_frag_t(Wt + kd, E2_);
      accp = wmma_bf16(a, bb, accp);
    }
    const float bias = b2[n0 + nn];
#pragma unroll
    for (int r = 0; r < 8; ++r) {
      const int row = r + 8 * hh;
      fbuf[row * D_ + n0 + nn] =
          accp[r] + bias + xf[((size_t)row0 + row) * D_ + n0 + nn];
    }
  }
  __syncthreads();

  // ---- parallel LayerNorm: 16 threads/row x 64 elems ----
  {
    const float* p = fbuf + rrow * D_ + rsub * 64;
    float ssum = 0.0f;
    for (int k = 0; k < 64; ++k) ssum += p[k];
    red[rrow * 16 + rsub] = ssum;
  }
  __syncthreads();
  if (tid < 16) {
    float mu = 0.0f;
    for (int k = 0; k < 16; ++k) mu += red[tid * 16 + k];
    mu_s[tid] = mu * (1.0f / D_);
  }
  __syncthreads();
  {
    const float mu = mu_s[rrow];
    const float* p = fbuf + rrow * D_ + rsub * 64;
    float ssum = 0.0f;
    for (int k = 0; k < 64; ++k) { float d = p[k] - mu; ssum += d * d; }
    red[rrow * 16 + rsub] = ssum;
  }
  __syncthreads();
  if (tid < 16) {
    float var = 0.0f;
    for (int k = 0; k < 16; ++k) var += red[tid * 16 + k];
    rs_s[tid] = rsqrtf(var * (1.0f / D_) + EPS_);
  }
  __syncthreads();

  for (int i = tid; i < 16 * D_; i += 256) {
    const int row = i >> 10;
    const int col = i & (D_ - 1);
    float v = (fbuf[i] - mu_s[row]) * rs_s[row] * g3[col] + be3[col];
    out[((size_t)row0 + row) * D_ + col] = v;
  }
}

// ---------------------------------------------------------------------------
// Host launcher
// ---------------------------------------------------------------------------
extern "C" void kernel_launch(void* const* d_in, const int* in_sizes, int n_in,
                              void* d_out, int out_size, void* d_ws, size_t ws_size,
                              hipStream_t stream) {
  const float* lags = (const float*)d_in[0];
  const float* W1   = (const float*)d_in[1];
  const float* b1   = (const float*)d_in[2];
  const float* W2   = (const float*)d_in[3];
  const float* b2   = (const float*)d_in[4];
  const float* g1   = (const float*)d_in[5];
  const float* be1  = (const float*)d_in[6];
  const float* g3   = (const float*)d_in[7];
  const float* be3  = (const float*)d_in[8];
  float* out = (float*)d_out;

  // workspace carve-up (~352 MB total)
  char* ws = (char*)d_ws;
  __bf16* lagsb = (__bf16*)ws; ws += (size_t)B_ * S_ * D_ * sizeof(__bf16);
  __bf16* lagsT = (__bf16*)ws; ws += (size_t)B_ * S_ * D_ * sizeof(__bf16);
  float*  xf    = (float*)ws;  ws += (size_t)B_ * S_ * D_ * sizeof(float);
  __bf16* xb    = (__bf16*)ws; ws += (size_t)B_ * S_ * D_ * sizeof(__bf16);
  __bf16* W1T   = (__bf16*)ws; ws += (size_t)D_ * E2_ * sizeof(__bf16);
  __bf16* W2T   = (__bf16*)ws; ws += (size_t)E2_ * D_ * sizeof(__bf16);

  const size_t nl = (size_t)B_ * S_ * D_;
  cast_bf16_kernel<<<4096, 256, 0, stream>>>(lags, lagsb, nl);

  {
    // per-batch transpose of lags: [S][D] -> [D][S] bf16
    dim3 grid((unsigned)(((size_t)S_ * D_ + 255) / 256), B_);
    transpose_cast_batched_kernel<<<grid, 256, 0, stream>>>(lags, lagsT, S_, D_);
  }
  transpose_cast_kernel<<<(unsigned)(((size_t)D_ * E2_ + 255) / 256), 256, 0, stream>>>(
      W1, W1T, D_, E2_);
  transpose_cast_kernel<<<(unsigned)(((size_t)E2_ * D_ + 255) / 256), 256, 0, stream>>>(
      W2, W2T, E2_, D_);

  // attention: grid (S/16 row-blocks, B batches)
  {
    dim3 grid(S_ / 16, B_);
    size_t smem = (size_t)16 * KT_ * sizeof(float)     // sc
                + (size_t)16 * D_ * sizeof(float)      // xbuf
                + 256 * sizeof(float)                  // red
                + 3 * 16 * sizeof(float)               // stats
                + (size_t)16 * D_ * sizeof(__bf16)     // qs
                + (size_t)16 * KT_ * sizeof(__bf16)    // pb
                + 64;
    attn_ln_kernel<<<grid, 256, smem, stream>>>(lags, lagsb, lagsT, g1, be1, xf, xb);
  }

  // fused FFN: one block per 16 rows of [B*S, D]
  {
    size_t smem = (size_t)16 * E2_ * sizeof(__bf16)    // hs
                + (size_t)16 * D_ * sizeof(float)      // fbuf
                + 256 * sizeof(float)                  // red
                + 2 * 16 * sizeof(float)               // stats
                + (size_t)16 * D_ * sizeof(__bf16)     // xs
                + 64;
    ffn_ln_kernel<<<(B_ * S_) / 16, 256, smem, stream>>>(
        xf, xb, W1T, b1, W2T, b2, g3, be3, out);
  }
}
